// EdgePooling_13348758356088
// MI455X (gfx1250) — matrix operand order repaired and checked by reading, not compiled, not run on hardware
//
#include <hip/hip_runtime.h>
#include <hip/hip_bf16.h>
#include <math.h>

typedef __attribute__((ext_vector_type(16))) _Float16 v16h;
typedef __attribute__((ext_vector_type(8)))  float    v8f;

#define N_NODES 8192
#define C_FEAT  128
#define E_EDGES 131072
#define ADD_SCORE 0.5f

// ---------- helpers ----------
__device__ __forceinline__ void atomicMaxF(float* addr, float val) {
  // monotonic-bits trick: ordered in int for >=0, reverse-ordered in uint for <0
  if (val >= 0.0f) atomicMax((int*)addr, __float_as_int(val));
  else             atomicMin((unsigned int*)addr, __float_as_uint(val));
}

// ---------- 1) edge scores via WMMA: raw = concat(x[src],x[dst]) . W + b ----------
// one wave (32 lanes) computes 16 edge scores; K=256 -> 8 chained v_wmma_f32_16x16x32_f16
__global__ void __launch_bounds__(256) ep_edge_score_wmma(
    const float* __restrict__ x, const int* __restrict__ src,
    const int* __restrict__ dst, const float* __restrict__ W,
    const float* __restrict__ bias, float* __restrict__ raw) {
  __shared__ _Float16 Wsh[2 * C_FEAT];
  int t = threadIdx.x;
  Wsh[t] = (_Float16)W[t];          // blockDim==256 == 2*C
  __syncthreads();

  int wave = t >> 5;
  int lane = t & 31;
  int tile = blockIdx.x * 8 + wave; // 16 edges per tile
  int e0   = tile * 16;

  int row = lane & 15;              // M index this lane feeds for A
  int kbA = (lane >> 4) * 8;        // A-frag K base within 32-chunk
  int kbB = (lane >> 4) * 16;       // unused: B frag indexes with kbA, like A
  int e = e0 + row;
  if (e >= E_EDGES) e = E_EDGES - 1;   // clamp, keep EXEC all-ones for WMMA
  int s = src[e], d = dst[e];
  const float* xs = x + (size_t)s * C_FEAT;
  const float* xd = x + (size_t)d * C_FEAT;

  v8f acc = {0.f, 0.f, 0.f, 0.f, 0.f, 0.f, 0.f, 0.f};
  #pragma unroll
  for (int c = 0; c < 8; ++c) {     // 8 chunks of K=32 cover 2C=256
    int base = c * 32;
    v16h a{}, bq{};
    #pragma unroll
    for (int i = 0; i < 16; ++i) {
      int kA = base + kbA + (i < 8 ? i : i + 8);   // 16-bit A 16x32 VGPR layout
      float av = (kA < C_FEAT) ? xs[kA] : xd[kA - C_FEAT];
      a[i] = (_Float16)av;
      int kB = base + kbA + (i < 8 ? i : i + 8);   // 16-bit B 32x16 VGPR layout
      bq[i] = Wsh[kB];                             // W replicated over all N cols
    }
    acc = __builtin_amdgcn_wmma_f32_16x16x32_f16(
        false, a, false, bq, (short)0, acc, false, false);
  }
  // C/D 16x16 f32 layout: lane 0 (N=0) holds M=0..7 in acc[0..7]; lane 16 M=8..15
  if ((lane & 15) == 0) {
    int mofs = (lane >> 4) * 8;
    float bb = bias[0];
    #pragma unroll
    for (int r = 0; r < 8; ++r) {
      int ee = e0 + mofs + r;
      if (ee < E_EDGES) raw[ee] = acc[r] + bb;
    }
  }
}

// ---------- 2) segment softmax over dst ----------
__global__ void ep_init_seg(float* segmax, float* denom) {
  int i = blockIdx.x * blockDim.x + threadIdx.x;
  if (i < N_NODES) { segmax[i] = -__builtin_inff(); denom[i] = 0.f; }
}
__global__ void ep_seg_max(const float* __restrict__ raw, const int* __restrict__ dst,
                           float* __restrict__ segmax) {
  int e = blockIdx.x * blockDim.x + threadIdx.x;
  if (e < E_EDGES) atomicMaxF(&segmax[dst[e]], raw[e]);
}
__global__ void ep_seg_exp(const float* __restrict__ raw, const int* __restrict__ dst,
                           const float* __restrict__ segmax, float* __restrict__ denom,
                           float* __restrict__ ex) {
  int e = blockIdx.x * blockDim.x + threadIdx.x;
  if (e < E_EDGES) {
    float v = expf(raw[e] - segmax[dst[e]]);
    ex[e] = v;
    atomicAdd(&denom[dst[e]], v);
  }
}
__global__ void ep_seg_norm(float* __restrict__ sco, const int* __restrict__ dst,
                            const float* __restrict__ denom) {
  int e = blockIdx.x * blockDim.x + threadIdx.x;
  if (e < E_EDGES) sco[e] = sco[e] / denom[dst[e]] + ADD_SCORE;
}

// ---------- 3) sort: key = (descending-orderable score)<<32 | idx ----------
__global__ void ep_pack_keys(const float* __restrict__ sco,
                             unsigned long long* __restrict__ keys) {
  int e = blockIdx.x * blockDim.x + threadIdx.x;
  if (e >= E_EDGES) return;
  unsigned u = __float_as_uint(sco[e]);
  u = (u & 0x80000000u) ? ~u : (u | 0x80000000u); // ascending-orderable bits
  u = ~u;                                          // descending
  keys[e] = ((unsigned long long)u << 32) | (unsigned)e;
}
__global__ void ep_bitonic_step(unsigned long long* __restrict__ keys, int k, int j) {
  int i = blockIdx.x * blockDim.x + threadIdx.x;
  if (i >= E_EDGES) return;
  int ixj = i ^ j;
  if (ixj > i) {
    unsigned long long a = keys[i], b = keys[ixj];
    bool up = ((i & k) == 0);
    if ((a > b) == up) { keys[i] = b; keys[ixj] = a; }
  }
}

// ---------- 4) sequential greedy contraction (inherently serial) ----------
__global__ void ep_greedy(const unsigned long long* __restrict__ keys,
                          const float* __restrict__ sco,
                          const int* __restrict__ src, const int* __restrict__ dst,
                          int* __restrict__ cluster, float* __restrict__ cs,
                          float* __restrict__ cscore,
                          unsigned char* __restrict__ freeArr, int* __restrict__ meta) {
  if (blockIdx.x != 0 || threadIdx.x != 0) return;
  for (int i = 0; i < N_NODES; ++i) { freeArr[i] = 1; cluster[i] = -1; cs[i] = 0.f; }
  int rank = 0;
  for (int e = 0; e < E_EDGES; ++e) {
    int idx = (int)(keys[e] & 0xffffffffull);
    int s = src[idx], t = dst[idx];
    if (freeArr[s] && freeArr[t]) {
      freeArr[s] = 0; freeArr[t] = 0;
      cluster[s] = rank; cluster[t] = rank;
      cs[rank] = sco[idx];
      ++rank;
    }
  }
  int num_sel = rank;
  for (int i = 0; i < N_NODES; ++i) if (freeArr[i]) cluster[i] = rank++;
  int num_clusters = rank;
  for (int col = 0; col < N_NODES; ++col)
    cscore[col] = (col < num_sel) ? cs[col] : ((col < num_clusters) ? 1.0f : 0.0f);
  meta[0] = num_sel; meta[1] = num_clusters;
}

// ---------- 5) pooled = segment_max(x, cluster); new_x ----------
__global__ void ep_fill_neginf(float* p, int count) {
  int i = blockIdx.x * blockDim.x + threadIdx.x;
  if (i < count) p[i] = -__builtin_inff();
}
__global__ void ep_pool_max(const float* __restrict__ x, const int* __restrict__ cluster,
                            float* __restrict__ pooled) {
  int i = blockIdx.x * blockDim.x + threadIdx.x;
  if (i >= N_NODES * C_FEAT) return;
  int node = i / C_FEAT, c = i % C_FEAT;
  atomicMaxF(&pooled[(size_t)cluster[node] * C_FEAT + c], x[i]);
}
__global__ void ep_new_x(const float* __restrict__ pooled, const float* __restrict__ cscore,
                         const int* __restrict__ meta, float* __restrict__ out_newx) {
  int i = blockIdx.x * blockDim.x + threadIdx.x;
  if (i >= N_NODES * C_FEAT) return;
  int col = i / C_FEAT;
  float v = (col < meta[1]) ? pooled[i] : 0.f;
  out_newx[i] = v * cscore[col];
}

// ---------- 6) attention / new_batch ----------
__global__ void ep_zero4(float* __restrict__ p, long long count4) {
  long long i = (long long)blockIdx.x * blockDim.x + threadIdx.x;
  long long stride = (long long)gridDim.x * blockDim.x;
  float4* p4 = (float4*)p;
  float4 z = make_float4(0.f, 0.f, 0.f, 0.f);
  for (; i < count4; i += stride) p4[i] = z;
}
__global__ void ep_zero(float* __restrict__ p, int count) {
  int i = blockIdx.x * blockDim.x + threadIdx.x;
  if (i < count) p[i] = 0.f;
}
__global__ void ep_att_scatter(const int* __restrict__ cluster,
                               const float* __restrict__ cscore,
                               float* __restrict__ att) {
  int i = blockIdx.x * blockDim.x + threadIdx.x;
  if (i >= N_NODES) return;
  int c = cluster[i];
  att[(size_t)i * N_NODES + c] = cscore[c];
}
__global__ void ep_batch_scatter(const int* __restrict__ batch,
                                 const int* __restrict__ cluster,
                                 float* __restrict__ nb) {
  int i = blockIdx.x * blockDim.x + threadIdx.x;
  if (i < N_NODES) nb[cluster[i]] = (float)batch[i];
}

// ---------- 7) coalesce(cluster[edge_index]) ----------
__global__ void ep_pack_edge_keys(const int* __restrict__ src, const int* __restrict__ dst,
                                  const int* __restrict__ cluster,
                                  unsigned long long* __restrict__ keys) {
  int e = blockIdx.x * blockDim.x + threadIdx.x;
  if (e < E_EDGES)
    keys[e] = (unsigned long long)cluster[src[e]] * N_NODES +
              (unsigned long long)cluster[dst[e]];
}
__global__ void ep_dedupe(const unsigned long long* __restrict__ in,
                          unsigned long long* __restrict__ out) {
  int e = blockIdx.x * blockDim.x + threadIdx.x;
  if (e >= E_EDGES) return;
  bool first = (e == 0) || (in[e] != in[e - 1]);
  out[e] = first ? in[e] : (unsigned long long)N_NODES * N_NODES;
}
__global__ void ep_emit_edges(const unsigned long long* __restrict__ keys,
                              float* __restrict__ out_ei) {
  int e = blockIdx.x * blockDim.x + threadIdx.x;
  if (e >= E_EDGES) return;
  unsigned long long k = keys[e];
  out_ei[e]            = (float)(k / N_NODES);
  out_ei[E_EDGES + e]  = (float)(k % N_NODES);
}
__global__ void ep_emit_tail(const int* __restrict__ cluster,
                             const float* __restrict__ cscore,
                             const int* __restrict__ meta,
                             float* __restrict__ out_cl, float* __restrict__ out_cs,
                             float* __restrict__ out_nc) {
  int i = blockIdx.x * blockDim.x + threadIdx.x;
  if (i < N_NODES) { out_cl[i] = (float)cluster[i]; out_cs[i] = cscore[i]; }
  if (i == 0) out_nc[0] = (float)meta[1];
}

// ---------- host ----------
extern "C" void kernel_launch(void* const* d_in, const int* in_sizes, int n_in,
                              void* d_out, int out_size, void* d_ws, size_t ws_size,
                              hipStream_t stream) {
  (void)in_sizes; (void)n_in; (void)out_size; (void)ws_size;
  const float* x     = (const float*)d_in[0];
  const int*   ei    = (const int*)d_in[1];
  const int*   src   = ei;
  const int*   dst   = ei + E_EDGES;
  const int*   batch = (const int*)d_in[2];
  const float* W     = (const float*)d_in[3];
  const float* bias  = (const float*)d_in[4];
  float* out = (float*)d_out;

  // output offsets (tuple flattened in return order)
  size_t off_newx = 0;
  size_t off_ei   = off_newx + (size_t)N_NODES * C_FEAT;
  size_t off_nb   = off_ei + 2 * (size_t)E_EDGES;
  size_t off_att  = off_nb + N_NODES;
  size_t off_cl   = off_att + (size_t)N_NODES * N_NODES;
  size_t off_cs   = off_cl + N_NODES;
  size_t off_nc   = off_cs + N_NODES;

  // workspace carve (256B aligned)
  char* wsp = (char*)d_ws;
  auto carve = [&](size_t bytes) {
    char* p = wsp; wsp += (bytes + 255) & ~(size_t)255; return p;
  };
  float* raw    = (float*)carve((size_t)E_EDGES * 4);
  float* sco    = (float*)carve((size_t)E_EDGES * 4);
  float* segmax = (float*)carve((size_t)N_NODES * 4);
  float* denom  = (float*)carve((size_t)N_NODES * 4);
  unsigned long long* keys  = (unsigned long long*)carve((size_t)E_EDGES * 8);
  unsigned long long* keys2 = (unsigned long long*)carve((size_t)E_EDGES * 8);
  int*   cluster = (int*)carve((size_t)N_NODES * 4);
  float* cs      = (float*)carve((size_t)N_NODES * 4);
  float* cscore  = (float*)carve((size_t)N_NODES * 4);
  unsigned char* freeArr = (unsigned char*)carve(N_NODES);
  float* pooled  = (float*)carve((size_t)N_NODES * C_FEAT * 4);
  int*   meta    = (int*)carve(64);

  const int TB = 256;
  const int gE = (E_EDGES + TB - 1) / TB;
  const int gN = (N_NODES + TB - 1) / TB;
  const int gNC = (N_NODES * C_FEAT + TB - 1) / TB;

  // 1) edge scores (WMMA): 16 edges/wave, 8 waves/block
  ep_edge_score_wmma<<<E_EDGES / (16 * 8), TB, 0, stream>>>(x, src, dst, W, bias, raw);

  // 2) segment softmax
  ep_init_seg<<<gN, TB, 0, stream>>>(segmax, denom);
  ep_seg_max<<<gE, TB, 0, stream>>>(raw, dst, segmax);
  ep_seg_exp<<<gE, TB, 0, stream>>>(raw, dst, segmax, denom, sco);
  ep_seg_norm<<<gE, TB, 0, stream>>>(sco, dst, denom);

  // 3) sort edges by descending score (stable via idx in low bits)
  ep_pack_keys<<<gE, TB, 0, stream>>>(sco, keys);
  for (int k = 2; k <= E_EDGES; k <<= 1)
    for (int j = k >> 1; j > 0; j >>= 1)
      ep_bitonic_step<<<gE, TB, 0, stream>>>(keys, k, j);

  // 4) sequential greedy matching
  ep_greedy<<<1, 1, 0, stream>>>(keys, sco, src, dst, cluster, cs, cscore, freeArr, meta);

  // 5) scatter-max pooling + new_x
  ep_fill_neginf<<<gNC, TB, 0, stream>>>(pooled, N_NODES * C_FEAT);
  ep_pool_max<<<gNC, TB, 0, stream>>>(x, cluster, pooled);
  ep_new_x<<<gNC, TB, 0, stream>>>(pooled, cscore, meta, out + off_newx);

  // 6) attention (256 MB zero + scatter) and new_batch
  ep_zero4<<<8192, TB, 0, stream>>>(out + off_att, (long long)N_NODES * N_NODES / 4);
  ep_att_scatter<<<gN, TB, 0, stream>>>(cluster, cscore, out + off_att);
  ep_zero<<<gN, TB, 0, stream>>>(out + off_nb, N_NODES);
  ep_batch_scatter<<<gN, TB, 0, stream>>>(batch, cluster, out + off_nb);

  // 7) coalesce new edge index: sort, dedupe, sort, emit
  ep_pack_edge_keys<<<gE, TB, 0, stream>>>(src, dst, cluster, keys);
  for (int k = 2; k <= E_EDGES; k <<= 1)
    for (int j = k >> 1; j > 0; j >>= 1)
      ep_bitonic_step<<<gE, TB, 0, stream>>>(keys, k, j);
  ep_dedupe<<<gE, TB, 0, stream>>>(keys, keys2);
  for (int k = 2; k <= E_EDGES; k <<= 1)
    for (int j = k >> 1; j > 0; j >>= 1)
      ep_bitonic_step<<<gE, TB, 0, stream>>>(keys2, k, j);
  ep_emit_edges<<<gE, TB, 0, stream>>>(keys2, out + off_ei);

  // 8) cluster / cluster_score / num_clusters
  ep_emit_tail<<<gN, TB, 0, stream>>>(cluster, cscore, meta,
                                      out + off_cl, out + off_cs, out + off_nc);
}
